// GINNet_80633716015159
// MI455X (gfx1250) — matrix operand order repaired and checked by reading, not compiled, not run on hardware
//
#include <hip/hip_runtime.h>

#define N_NODES 50000
#define N_EDGES 1600000
#define DIM 128
#define NLAYERS 4
#define BSTRIDE 136  // padded bf16 row stride for B in LDS (272B = 68 dwords -> conflict-free b128 frags)

typedef __attribute__((ext_vector_type(16))) __bf16 v16bf;
typedef __attribute__((ext_vector_type(8)))  float  v8f;

__device__ __forceinline__ void atomicAddF(float* p, float v) {
  unsafeAtomicAdd(p, v);   // global_atomic_add_f32, agent scope
}

// A-fragment K map used by this code: contiguous, element s of lane half h -> k = 16*half + s + 32*ks
__device__ __forceinline__ int kmapA(int e, int half, int ks) {
  return 2 * e + 16 * half + 32 * ks;
}

// -------- zero scratch (neigh buffer + 6 stat accumulators) --------
__global__ void zero_kernel(float* __restrict__ a, float* __restrict__ stats, int nA) {
  int idx = blockIdx.x * blockDim.x + threadIdx.x;
  if (idx < nA) a[idx] = 0.0f;
  else if (idx - nA < 6 * DIM) stats[idx - nA] = 0.0f;
}

// -------- edge scatter-add: one wave per edge, lane = one float4 --------
__global__ void agg_kernel(const float* __restrict__ h, const int* __restrict__ src,
                           const int* __restrict__ dst, float* __restrict__ neigh, int E) {
  int gid = blockIdx.x * blockDim.x + threadIdx.x;
  int e = gid >> 5;
  if (e >= E) return;
  int lane = threadIdx.x & 31;
  int s = src[e], d = dst[e];
  const float4* hv = (const float4*)(h + (size_t)s * DIM);
  float4 v = hv[lane];
  float* np = neigh + (size_t)d * DIM + lane * 4;
  atomicAddF(np + 0, v.x);
  atomicAddF(np + 1, v.y);
  atomicAddF(np + 2, v.z);
  atomicAddF(np + 3, v.w);
}

// B fragment: 16 contiguous bf16 of LDS row n starting at K = half*16 + 32*ks
__device__ __forceinline__ v16bf load_bfrag(const unsigned short* Blds, int n, int half, int ks) {
  union { v16bf bf; uint4 q[2]; } bb;
  const uint4* bp = (const uint4*)&Blds[n * BSTRIDE + half * 16 + 32 * ks];
  bb.q[0] = bp[0];
  bb.q[1] = bp[1];
  return bb.bf;
}

// A fragment builder (contiguous K map), f32 -> bf16, with fused input transform
template <int MODE>
__device__ __forceinline__ v16bf build_afrag(const float* arow, const float* a2row,
                                             const float* trS, const float* trB,
                                             float alpha, int half, int ks) {
  v16bf afrag;
#pragma unroll
  for (int e = 0; e < 8; ++e) {
    int k = kmapA(e, half, ks);
    float x0 = arow[k], x1 = arow[k + 1];
    if (MODE == 1) { x0 = alpha * x0 + a2row[k]; x1 = alpha * x1 + a2row[k + 1]; }
    if (MODE == 2) {
      x0 = fmaxf(trS[k] * x0 + trB[k], 0.0f);
      x1 = fmaxf(trS[k + 1] * x1 + trB[k + 1], 0.0f);
    }
    afrag[2 * e]     = (__bf16)x0;
    afrag[2 * e + 1] = (__bf16)x1;
  }
  return afrag;
}

#define WMMA_BF16(A, B, C) \
  __builtin_amdgcn_wmma_f32_16x16x32_bf16(false, (A), false, (B), (short)0, (C), false, false)

// -------- WMMA GEMM: out[N,128] = f(A)[N,128] @ W[128,128] + bias --------
// Four waves per 16-row M-tile; each wave owns 32 output columns (2 n-tiles),
// so its 8 B-fragments (64 VGPRs) stay register-resident across the tile loop.
// MODE 0: f(A)=A (embed, no stats)
// MODE 1: f(A)=(1+eps)*A + A2 (GIN combine), collect column stats of output
// MODE 2: f(A)=relu(trS[k]*A + trB[k]) (BN1+relu on input), collect stats
template <int MODE>
__global__ __launch_bounds__(256, 1)
void gemm128(const float* __restrict__ A, const float* __restrict__ A2,
             const float* __restrict__ trS, const float* __restrict__ trB,
             const float* __restrict__ epsp, const float* __restrict__ W,
             const float* __restrict__ bias, float* __restrict__ out,
             float* __restrict__ colsum, float* __restrict__ colsumsq, int nrows) {
  __shared__ unsigned short Blds[DIM * BSTRIDE];  // bf16 bits, layout [n][k], padded rows
  for (int j = threadIdx.x; j < DIM * DIM; j += blockDim.x) {
    int n = j >> 7, k = j & 127;
    union { __bf16 b; unsigned short u; } cv;
    cv.b = (__bf16)W[k * DIM + n];
    Blds[n * BSTRIDE + k] = cv.u;
  }
  __syncthreads();

  float alpha = 1.0f;
  if (MODE == 1) alpha = 1.0f + epsp[0];

  int wave   = blockIdx.x * (blockDim.x >> 5) + (threadIdx.x >> 5);
  int nWaves = gridDim.x * (blockDim.x >> 5);   // multiple of 4 -> nquarter is per-wave constant
  int lane   = threadIdx.x & 31;
  int half   = lane >> 4;
  int mlane  = lane & 15;
  int nbase  = (wave & 3) * 32;   // which 32-column quarter this wave owns
  int n0 = nbase + mlane;
  int n1 = nbase + 16 + mlane;

  // ---- preload this wave's 8 B fragments into named registers (64 VGPRs) ----
  v16bf b00 = load_bfrag(Blds, n0, half, 0);
  v16bf b01 = load_bfrag(Blds, n1, half, 0);
  v16bf b10 = load_bfrag(Blds, n0, half, 1);
  v16bf b11 = load_bfrag(Blds, n1, half, 1);
  v16bf b20 = load_bfrag(Blds, n0, half, 2);
  v16bf b21 = load_bfrag(Blds, n1, half, 2);
  v16bf b30 = load_bfrag(Blds, n0, half, 3);
  v16bf b31 = load_bfrag(Blds, n1, half, 3);

  int mtiles = nrows >> 4;  // 50000/16 = 3125 exact
  for (int tile = (wave >> 2); tile < mtiles; tile += (nWaves >> 2)) {
    int row = tile * 16 + mlane;
    const float* arow  = A + (size_t)row * DIM;
    const float* a2row = (MODE == 1) ? (A2 + (size_t)row * DIM) : A;

    v8f acc0 = (v8f){0, 0, 0, 0, 0, 0, 0, 0};
    v8f acc1 = (v8f){0, 0, 0, 0, 0, 0, 0, 0};

    v16bf a0 = build_afrag<MODE>(arow, a2row, trS, trB, alpha, half, 0);
    acc0 = WMMA_BF16(a0, b00, acc0);
    acc1 = WMMA_BF16(a0, b01, acc1);
    v16bf a1 = build_afrag<MODE>(arow, a2row, trS, trB, alpha, half, 1);
    acc0 = WMMA_BF16(a1, b10, acc0);
    acc1 = WMMA_BF16(a1, b11, acc1);
    v16bf a2 = build_afrag<MODE>(arow, a2row, trS, trB, alpha, half, 2);
    acc0 = WMMA_BF16(a2, b20, acc0);
    acc1 = WMMA_BF16(a2, b21, acc1);
    v16bf a3 = build_afrag<MODE>(arow, a2row, trS, trB, alpha, half, 3);
    acc0 = WMMA_BF16(a3, b30, acc0);
    acc1 = WMMA_BF16(a3, b31, acc1);

    // ---- epilogue: bias, store, fused BN stat atomics ----
#pragma unroll
    for (int nt = 0; nt < 2; ++nt) {
      int n = nbase + nt * 16 + mlane;
      v8f accv = nt ? acc1 : acc0;
      float bv = bias[n];
      float s = 0.0f, sq = 0.0f;
#pragma unroll
      for (int r = 0; r < 8; ++r) {
        int m = r + 8 * half;
        float v = accv[r] + bv;
        out[(size_t)(tile * 16 + m) * DIM + n] = v;
        s += v;
        sq += v * v;
      }
      if (MODE != 0) {
        s  += __shfl_xor(s, 16);
        sq += __shfl_xor(sq, 16);
        if (lane < 16) {
          atomicAddF(&colsum[n], s);
          atomicAddF(&colsumsq[n], sq);
        }
      }
    }
  }
}

// -------- fold BN into per-column scale/shift --------
__global__ void bn_finalize(const float* __restrict__ colsum, const float* __restrict__ colsumsq,
                            const float* __restrict__ g, const float* __restrict__ be,
                            float* __restrict__ sa, float* __restrict__ sb, float invN) {
  int c = threadIdx.x;
  float mu  = colsum[c] * invN;
  float var = colsumsq[c] * invN - mu * mu;
  float a = g[c] * rsqrtf(var + 1e-5f);
  sa[c] = a;
  sb[c] = be[c] - a * mu;
}

// -------- u = relu(bn2(z)), collect column stats of u --------
__global__ void relu_bn_stats(const float* __restrict__ z, const float* __restrict__ sa,
                              const float* __restrict__ sb, float* __restrict__ u,
                              float* __restrict__ colsum, float* __restrict__ colsumsq, int nrows) {
  int col = threadIdx.x & 127;
  int rsub = threadIdx.x >> 7;
  int base = blockIdx.x * 64;
  float a = sa[col], b = sb[col];
  float s = 0.0f, sq = 0.0f;
  for (int i = 0; i < 32; ++i) {
    int row = base + rsub + 2 * i;
    if (row < nrows) {
      float v = fmaxf(a * z[(size_t)row * DIM + col] + b, 0.0f);
      u[(size_t)row * DIM + col] = v;
      s += v;
      sq += v * v;
    }
  }
  atomicAddF(&colsum[col], s);
  atomicAddF(&colsumsq[col], sq);
}

// -------- h_new = h_in + relu(bn3(u)) --------
__global__ void final_kernel(const float* hin, const float* __restrict__ u,
                             const float* __restrict__ sa, const float* __restrict__ sb,
                             float* out, int total) {
  int idx = blockIdx.x * blockDim.x + threadIdx.x;
  if (idx >= total) return;
  int col = idx & 127;
  out[idx] = hin[idx] + fmaxf(sa[col] * u[idx] + sb[col], 0.0f);
}

extern "C" void kernel_launch(void* const* d_in, const int* in_sizes, int n_in,
                              void* d_out, int out_size, void* d_ws, size_t ws_size,
                              hipStream_t stream) {
  const float* h0   = (const float*)d_in[0];
  const int*   src  = (const int*)d_in[1];
  const int*   dst  = (const int*)d_in[2];
  const float* Wemb = (const float*)d_in[3];
  const float* bemb = (const float*)d_in[4];
  const float* epsA = (const float*)d_in[5];
  const float* W1   = (const float*)d_in[6];
  const float* b1   = (const float*)d_in[7];
  const float* g1   = (const float*)d_in[8];
  const float* be1  = (const float*)d_in[9];
  const float* W2   = (const float*)d_in[10];
  const float* b2   = (const float*)d_in[11];
  const float* ga   = (const float*)d_in[12];
  const float* ba   = (const float*)d_in[13];
  const float* gl   = (const float*)d_in[14];
  const float* bl   = (const float*)d_in[15];

  float* ws = (float*)d_ws;
  const size_t ND = (size_t)N_NODES * DIM;
  float* h_cur = ws;            // current node features (kept as residual input)
  float* bufB  = ws + ND;       // neigh, then z
  float* bufC  = ws + 2 * ND;   // y, then u
  float* st    = ws + 3 * ND;   // stats + folded BN params
  float *cs1 = st,        *cq1 = st + 128, *cs2 = st + 256, *cq2 = st + 384;
  float *cs3 = st + 512,  *cq3 = st + 640;
  float *sa1 = st + 768,  *sb1 = st + 896, *sa2 = st + 1024, *sb2 = st + 1152;
  float *sa3 = st + 1280, *sb3 = st + 1408;

  dim3 blk(256);
  const int gemmBlocks = 120;                       // 960 waves (mult of 4) over 12500 jobs
  const int zeroBlocks = ((int)ND + 6 * DIM + 255) / 256;
  const int aggBlocks  = (N_EDGES * 32) / 256;      // one wave per edge
  const int rbBlocks   = (N_NODES + 63) / 64;
  const int fnBlocks   = ((int)ND + 255) / 256;
  const float invN = 1.0f / (float)N_NODES;

  // embedding: h = h0 @ W_embed + b_embed
  gemm128<0><<<gemmBlocks, blk, 0, stream>>>(h0, nullptr, nullptr, nullptr, nullptr,
                                             Wemb, bemb, h_cur, nullptr, nullptr, N_NODES);

  for (int i = 0; i < NLAYERS; ++i) {
    zero_kernel<<<zeroBlocks, blk, 0, stream>>>(bufB, st, (int)ND);
    agg_kernel<<<aggBlocks, blk, 0, stream>>>(h_cur, src, dst, bufB, N_EDGES);
    // y = ((1+eps)*h + neigh) @ W1 + b1   (+ stats of y)
    gemm128<1><<<gemmBlocks, blk, 0, stream>>>(h_cur, bufB, nullptr, nullptr, epsA + i,
                                               W1 + (size_t)i * DIM * DIM, b1 + i * DIM,
                                               bufC, cs1, cq1, N_NODES);
    bn_finalize<<<1, DIM, 0, stream>>>(cs1, cq1, g1 + i * DIM, be1 + i * DIM, sa1, sb1, invN);
    // z = relu(bn1(y)) @ W2 + b2   (+ stats of z)
    gemm128<2><<<gemmBlocks, blk, 0, stream>>>(bufC, nullptr, sa1, sb1, nullptr,
                                               W2 + (size_t)i * DIM * DIM, b2 + i * DIM,
                                               bufB, cs2, cq2, N_NODES);
    bn_finalize<<<1, DIM, 0, stream>>>(cs2, cq2, ga + i * DIM, ba + i * DIM, sa2, sb2, invN);
    // u = relu(bn2(z))   (+ stats of u)
    relu_bn_stats<<<rbBlocks, blk, 0, stream>>>(bufB, sa2, sb2, bufC, cs3, cq3, N_NODES);
    bn_finalize<<<1, DIM, 0, stream>>>(cs3, cq3, gl + i * DIM, bl + i * DIM, sa3, sb3, invN);
    // h = h_in + relu(bn3(u))
    float* outp = (i == NLAYERS - 1) ? (float*)d_out : h_cur;
    final_kernel<<<fnBlocks, blk, 0, stream>>>(h_cur, bufC, sa3, sb3, outp, (int)ND);
  }
}
